// BRDFGatherModel_28810640622012
// MI455X (gfx1250) — matrix operand order repaired and checked by reading, not compiled, not run on hardware
//
#include <hip/hip_runtime.h>
#include <hip/hip_bf16.h>

typedef __attribute__((ext_vector_type(16))) _Float16 v16h;
typedef __attribute__((ext_vector_type(8)))  _Float16 v8h;
typedef __attribute__((ext_vector_type(8)))  float    v8f;

#define NRAYS 32768
#define FSTR  72      // feat row stride (halves): K=39 padded to 64, +8 pad, 144B (16B-mult)
#define ASTR  136     // act  row stride (halves): K=128, +8 pad, 272B (16B-mult)
#define NFSTR 40      // naive feat stride: K=25 padded to 32, 80B (16B-mult)

union FragU { v16h v; v8h h[2]; };

// A-matrix 16x32 f16 fragment (ISA 7.12.2): lane holds M=lane%16; g=lane/16.
// VGPR0-3 = halves K = 32*ks + 8g .. +7 ; VGPR4-7 = K = 32*ks + 16 + 8g .. +7
__device__ __forceinline__ v16h load_frag_a(const _Float16* p, int g) {
    FragU u;
    u.h[0] = *(const v8h*)(p + 8 * g);
    u.h[1] = *(const v8h*)(p + 16 + 8 * g);
    return u.v;
}
// B-matrix 32x16 f16 fragment: lane holds N=lane%16; halves 0-7 K = 8g..+7, halves 8-15 K = 16+8g..+7
__device__ __forceinline__ v16h load_frag_b(const _Float16* p, int g) {
    FragU u;
    u.h[0] = *(const v8h*)(p + 8 * g);
    u.h[1] = *(const v8h*)(p + 16 + 8 * g);
    return u.v;
}
__device__ __forceinline__ v8f wmma16(v16h a, v16h b, v8f c) {
    return __builtin_amdgcn_wmma_f32_16x16x32_f16(false, a, false, b, (short)0, c,
                                                  false, false);
}

// Convert f32 weight [Kin][Nout] (row-major) -> f16 transposed [Npad][Kpad], zero-padded.
__global__ void convert_wT(const float* __restrict__ src, _Float16* __restrict__ dst,
                           int Kin, int Nout, int Kpad, int Npad) {
    int i = blockIdx.x * blockDim.x + threadIdx.x;
    if (i >= Npad * Kpad) return;
    int n = i / Kpad, k = i % Kpad;
    float v = (n < Nout && k < Kin) ? src[k * Nout + n] : 0.0f;
    dst[i] = (_Float16)v;
}

// ---------------- main fused NeRF kernel: 2 rays x 64 padded samples per block ----------
__global__ void __launch_bounds__(256)
nerf_main(const float* __restrict__ ofs, const float* __restrict__ results,
          const float* __restrict__ noise,
          const _Float16* __restrict__ ffW1, const _Float16* __restrict__ ffW2,
          const _Float16* __restrict__ ffW3,
          const _Float16* __restrict__ wfW1, const _Float16* __restrict__ wfW2,
          const _Float16* __restrict__ wfW3,
          const float* __restrict__ ffb1, const float* __restrict__ ffb2,
          const float* __restrict__ ffb3,
          const float* __restrict__ wfb1, const float* __restrict__ wfb2,
          const float* __restrict__ wfb3,
          float* __restrict__ out) {
    __shared__ __align__(16) _Float16 feat[128 * FSTR];
    __shared__ __align__(16) _Float16 act[128 * ASTR];
    __shared__ float pos_s[128 * 3];
    __shared__ float col_s[128 * 4];
    __shared__ float sig_s[128];
    __shared__ float alpha_s[128];

    const int tid = threadIdx.x;

    // warm WGP$/L1 for the L2-resident weights this block will stream
    if (tid == 0) {
        __builtin_prefetch(ffW1, 0, 0);
        __builtin_prefetch(ffW2, 0, 0);
        __builtin_prefetch(ffW3, 0, 0);
        __builtin_prefetch(wfW1, 0, 0);
        __builtin_prefetch(wfW2, 0, 0);
        __builtin_prefetch(wfW3, 0, 0);
    }

    // ---- phase 1: ray math + positional encoding (f16 into LDS) ----
    if (tid < 128) {
        const int pt  = tid;
        const int ray = blockIdx.x * 2 + (pt >> 6);
        const int s   = pt & 63;
        float o[3], r[3];
#pragma unroll
        for (int i = 0; i < 3; i++) {
            o[i] = ofs[ray * 3 + i] + noise[ray * 3 + i] * 1e-3f;
            r[i] = results[ray * 3 + i];
        }
        float tmin = fmaxf(fmaxf(r[0] - o[0], r[1] - o[1]), fmaxf(r[2] - o[2], 1e-4f));
        float l0 = sqrtf(r[0] * r[0] + r[1] * r[1] + r[2] * r[2]);
        float dm = (o[0] + o[1] + o[2]) * (1.0f / 3.0f);
        float inv_near = l0 / (tmin + dm);
        float inv_far  = l0 / (5.0f + dm);
        float u  = (float)s * (1.0f / 49.0f);
        float sv = (inv_near - inv_far) * u;
        float ts = l0 / (inv_near - sv) - dm;
        float p[3];
#pragma unroll
        for (int i = 0; i < 3; i++) p[i] = (s < 50) ? r[i] / (ts + o[i]) : 0.0f;
        pos_s[pt * 3 + 0] = p[0];
        pos_s[pt * 3 + 1] = p[1];
        pos_s[pt * 3 + 2] = p[2];
        _Float16* f = &feat[pt * FSTR];
        f[0] = (_Float16)p[0]; f[1] = (_Float16)p[1]; f[2] = (_Float16)p[2];
#pragma unroll
        for (int m = 0; m < 6; m++) {
            float sc = (float)(1 << m);
#pragma unroll
            for (int i = 0; i < 3; i++) {
                f[3 + 6 * m + i]     = (_Float16)__sinf(p[i] * sc);
                f[3 + 6 * m + 3 + i] = (_Float16)__cosf(p[i] * sc);
            }
        }
#pragma unroll
        for (int k = 39; k < FSTR; k++) f[k] = (_Float16)0.0f;
    }
    __syncthreads();

    const int wave = tid >> 5;
    const int lane = tid & 31;
    const int g    = lane >> 4;
    const int n0   = lane & 15;
    const int mrow = wave * 16;          // each wave owns one 16-row M-tile
    const int arow = mrow + (lane & 15); // A-fragment row for this lane
    const v8f vzero = {0.f, 0.f, 0.f, 0.f, 0.f, 0.f, 0.f, 0.f};

    auto run_net = [&](const _Float16* W1, const _Float16* W2, const _Float16* W3,
                       const float* b1, const float* b2, const float* b3, int mode) {
        v8f acc[8];
        // ---- layer 1: [16 x 64] @ [64 x 128]
#pragma unroll
        for (int n = 0; n < 8; n++) acc[n] = vzero;
#pragma unroll
        for (int ks = 0; ks < 2; ks++) {
            v16h a = load_frag_a(&feat[arow * FSTR + 32 * ks], g);
#pragma unroll
            for (int n = 0; n < 8; n++) {
                v16h b = load_frag_b(&W1[(n * 16 + n0) * 64 + 32 * ks], g);
                acc[n] = wmma16(a, b, acc[n]);
            }
        }
#pragma unroll
        for (int n = 0; n < 8; n++) {
            float bias = b1[n * 16 + n0];
#pragma unroll
            for (int rr = 0; rr < 8; rr++) {
                int pt = mrow + rr + 8 * g;   // D layout: M = vgpr + 8*(lane>=16)
                float v = fmaxf(acc[n][rr] + bias, 0.0f);
                act[pt * ASTR + n * 16 + n0] = (_Float16)v;
            }
        }
        // ---- layer 2: [16 x 128] @ [128 x 128]
#pragma unroll
        for (int n = 0; n < 8; n++) acc[n] = vzero;
#pragma unroll
        for (int ks = 0; ks < 4; ks++) {
            v16h a = load_frag_a(&act[arow * ASTR + 32 * ks], g);
#pragma unroll
            for (int n = 0; n < 8; n++) {
                v16h b = load_frag_b(&W2[(n * 16 + n0) * 128 + 32 * ks], g);
                acc[n] = wmma16(a, b, acc[n]);
            }
        }
#pragma unroll
        for (int n = 0; n < 8; n++) {
            float bias = b2[n * 16 + n0];
#pragma unroll
            for (int rr = 0; rr < 8; rr++) {
                int pt = mrow + rr + 8 * g;
                float v = fmaxf(acc[n][rr] + bias, 0.0f);
                act[pt * ASTR + n * 16 + n0] = (_Float16)v;
            }
        }
        // ---- layer 3: [16 x 128] @ [128 x 16] (output cols padded to 16)
        v8f a3 = vzero;
#pragma unroll
        for (int ks = 0; ks < 4; ks++) {
            v16h a = load_frag_a(&act[arow * ASTR + 32 * ks], g);
            v16h b = load_frag_b(&W3[n0 * 128 + 32 * ks], g);
            a3 = wmma16(a, b, a3);
        }
        if (mode == 0) {            // color net -> 3 columns
            if (n0 < 3) {
                float bias = b3[n0];
#pragma unroll
                for (int rr = 0; rr < 8; rr++) {
                    int pt = mrow + rr + 8 * g;
                    col_s[pt * 4 + n0] = a3[rr] + bias;
                }
            }
        } else {                    // sigma net -> 1 column
            if (n0 == 0) {
                float bias = b3[0];
#pragma unroll
                for (int rr = 0; rr < 8; rr++) {
                    int pt = mrow + rr + 8 * g;
                    sig_s[pt] = a3[rr] + bias;
                }
            }
        }
    };
    run_net(ffW1, ffW2, ffW3, ffb1, ffb2, ffb3, 0);  // color field
    run_net(wfW1, wfW2, wfW3, wfb1, wfb2, wfb3, 1);  // weight (sigma) field
    __syncthreads();

    // ---- alpha: parallel transcendentals, one thread per sample ----
    if (tid < 128) {
        float sg = log1pf(__expf(sig_s[tid] - 10.0f));   // softplus(x - 10)
        alpha_s[tid] = 1.0f - __expf(-sg);
    }
    __syncthreads();

    // ---- serial cumprod scan per ray: FMAs only ----
    if (tid < 2) {
        const int ray = blockIdx.x * 2 + tid;
        float T = 1.0f, wsum = 0.0f;
        float fx[3] = {0.f, 0.f, 0.f}, fc[3] = {0.f, 0.f, 0.f};
        for (int s = 0; s < 50; s++) {
            const int pt = tid * 64 + s;
            float alpha = alpha_s[pt];
            float w     = alpha * T;
            T *= (1.0f - alpha + 1e-10f);
#pragma unroll
            for (int i = 0; i < 3; i++) {
                fx[i] += pos_s[pt * 3 + i] * w;
                fc[i] += col_s[pt * 4 + i] * w;
            }
            wsum += w;
        }
#pragma unroll
        for (int i = 0; i < 3; i++) {
            out[ray * 3 + i]              = fx[i];  // final_x
            out[4 * NRAYS + ray * 3 + i]  = fc[i];  // final_c
        }
        out[7 * NRAYS + ray] = wsum;                // weight sum
    }
}

// ---------------- naive branch: WMMA MLP 25->128->128->6, then tiny per-thread MLP ----
__global__ void __launch_bounds__(256)
naive_kern(const float* __restrict__ x,
           const _Float16* __restrict__ W1, const _Float16* __restrict__ W2,
           const _Float16* __restrict__ W3,
           const float* __restrict__ b1, const float* __restrict__ b2,
           const float* __restrict__ b3,
           const float* __restrict__ mW1, const float* __restrict__ mb1,
           const float* __restrict__ mW2, const float* __restrict__ mb2,
           const float* __restrict__ mW3, const float* __restrict__ mb3,
           float* __restrict__ out) {
    __shared__ __align__(16) _Float16 featn[128 * NFSTR];
    __shared__ __align__(16) _Float16 act[128 * ASTR];
    __shared__ float rho_s[128 * 8];

    const int tid = threadIdx.x;
    if (tid == 0) {
        __builtin_prefetch(W1, 0, 0);
        __builtin_prefetch(W2, 0, 0);
        __builtin_prefetch(W3, 0, 0);
    }
    if (tid < 128) {
        const int ray = blockIdx.x * 128 + tid;
        float xv = x[ray];
        _Float16* f = &featn[tid * NFSTR];
        f[0] = (_Float16)xv;
#pragma unroll
        for (int m = 0; m < 12; m++) {
            float sc = (float)(1 << m);
            f[1 + 2 * m] = (_Float16)__sinf(xv * sc);
            f[2 + 2 * m] = (_Float16)__cosf(xv * sc);
        }
#pragma unroll
        for (int k = 25; k < NFSTR; k++) f[k] = (_Float16)0.0f;
    }
    __syncthreads();

    const int wave = tid >> 5;
    const int lane = tid & 31;
    const int g    = lane >> 4;
    const int n0   = lane & 15;
    const int mrow = wave * 16;
    const int arow = mrow + (lane & 15);
    const v8f vzero = {0.f, 0.f, 0.f, 0.f, 0.f, 0.f, 0.f, 0.f};

    v8f acc[8];
    // layer 1: K=32 (one k-step)
#pragma unroll
    for (int n = 0; n < 8; n++) acc[n] = vzero;
    {
        v16h a = load_frag_a(&featn[arow * NFSTR], g);
#pragma unroll
        for (int n = 0; n < 8; n++) {
            v16h b = load_frag_b(&W1[(n * 16 + n0) * 32], g);
            acc[n] = wmma16(a, b, acc[n]);
        }
    }
#pragma unroll
    for (int n = 0; n < 8; n++) {
        float bias = b1[n * 16 + n0];
#pragma unroll
        for (int rr = 0; rr < 8; rr++) {
            int pt = mrow + rr + 8 * g;
            act[pt * ASTR + n * 16 + n0] = (_Float16)fmaxf(acc[n][rr] + bias, 0.0f);
        }
    }
    // layer 2: K=128
#pragma unroll
    for (int n = 0; n < 8; n++) acc[n] = vzero;
#pragma unroll
    for (int ks = 0; ks < 4; ks++) {
        v16h a = load_frag_a(&act[arow * ASTR + 32 * ks], g);
#pragma unroll
        for (int n = 0; n < 8; n++) {
            v16h b = load_frag_b(&W2[(n * 16 + n0) * 128 + 32 * ks], g);
            acc[n] = wmma16(a, b, acc[n]);
        }
    }
#pragma unroll
    for (int n = 0; n < 8; n++) {
        float bias = b2[n * 16 + n0];
#pragma unroll
        for (int rr = 0; rr < 8; rr++) {
            int pt = mrow + rr + 8 * g;
            act[pt * ASTR + n * 16 + n0] = (_Float16)fmaxf(acc[n][rr] + bias, 0.0f);
        }
    }
    // layer 3: K=128 -> 6 outputs (padded 16), sigmoid
    v8f a3 = vzero;
#pragma unroll
    for (int ks = 0; ks < 4; ks++) {
        v16h a = load_frag_a(&act[arow * ASTR + 32 * ks], g);
        v16h b = load_frag_b(&W3[n0 * 128 + 32 * ks], g);
        a3 = wmma16(a, b, a3);
    }
    if (n0 < 6) {
        float bias = b3[n0];
#pragma unroll
        for (int rr = 0; rr < 8; rr++) {
            int pt = mrow + rr + 8 * g;
            float v = a3[rr] + bias;
            rho_s[pt * 8 + n0] = 1.0f / (1.0f + __expf(-v));
        }
    }
    __syncthreads();

    // tiny MLP 6->32->32->4, take column 3; one thread per ray
    if (tid < 128) {
        float r6[6];
#pragma unroll
        for (int j = 0; j < 6; j++) r6[j] = rho_s[tid * 8 + j];
        float h1[32];
#pragma unroll
        for (int o = 0; o < 32; o++) {
            float a = mb1[o];
#pragma unroll
            for (int j = 0; j < 6; j++) a += r6[j] * mW1[j * 32 + o];
            h1[o] = fmaxf(a, 0.0f);
        }
        float out3 = mb3[3];
        for (int o2 = 0; o2 < 32; o2++) {
            float a = mb2[o2];
#pragma unroll
            for (int o1 = 0; o1 < 32; o1++) a += h1[o1] * mW2[o1 * 32 + o2];
            out3 += fmaxf(a, 0.0f) * mW3[o2 * 4 + 3];
        }
        out[3 * NRAYS + blockIdx.x * 128 + tid] = out3;   // naive2
    }
}

extern "C" void kernel_launch(void* const* d_in, const int* in_sizes, int n_in,
                              void* d_out, int out_size, void* d_ws, size_t ws_size,
                              hipStream_t stream) {
    const float* x       = (const float*)d_in[0];
    const float* ofs     = (const float*)d_in[1];
    const float* results = (const float*)d_in[2];
    const float* noise   = (const float*)d_in[3];
    const float* wf_W1 = (const float*)d_in[4];  const float* wf_b1 = (const float*)d_in[5];
    const float* wf_W2 = (const float*)d_in[6];  const float* wf_b2 = (const float*)d_in[7];
    const float* wf_W3 = (const float*)d_in[8];  const float* wf_b3 = (const float*)d_in[9];
    const float* ff_W1 = (const float*)d_in[10]; const float* ff_b1 = (const float*)d_in[11];
    const float* ff_W2 = (const float*)d_in[12]; const float* ff_b2 = (const float*)d_in[13];
    const float* ff_W3 = (const float*)d_in[14]; const float* ff_b3 = (const float*)d_in[15];
    const float* nf_W1 = (const float*)d_in[16]; const float* nf_b1 = (const float*)d_in[17];
    const float* nf_W2 = (const float*)d_in[18]; const float* nf_b2 = (const float*)d_in[19];
    const float* nf_W3 = (const float*)d_in[20]; const float* nf_b3 = (const float*)d_in[21];
    const float* nm_W1 = (const float*)d_in[22]; const float* nm_b1 = (const float*)d_in[23];
    const float* nm_W2 = (const float*)d_in[24]; const float* nm_b2 = (const float*)d_in[25];
    const float* nm_W3 = (const float*)d_in[26]; const float* nm_b3 = (const float*)d_in[27];

    _Float16* ws = (_Float16*)d_ws;
    size_t off = 0;
    _Float16* ffW1T = ws + off; off += 128 * 64;
    _Float16* ffW2T = ws + off; off += 128 * 128;
    _Float16* ffW3T = ws + off; off += 16 * 128;
    _Float16* wfW1T = ws + off; off += 128 * 64;
    _Float16* wfW2T = ws + off; off += 128 * 128;
    _Float16* wfW3T = ws + off; off += 16 * 128;
    _Float16* nfW1T = ws + off; off += 128 * 32;
    _Float16* nfW2T = ws + off; off += 128 * 128;
    _Float16* nfW3T = ws + off; off += 16 * 128;

    auto conv = [&](const float* s, _Float16* d, int Kin, int Nout, int Kpad, int Npad) {
        int total = Npad * Kpad;
        convert_wT<<<(total + 255) / 256, 256, 0, stream>>>(s, d, Kin, Nout, Kpad, Npad);
    };
    conv(ff_W1, ffW1T, 39, 128, 64, 128);
    conv(ff_W2, ffW2T, 128, 128, 128, 128);
    conv(ff_W3, ffW3T, 128, 3, 128, 16);
    conv(wf_W1, wfW1T, 39, 128, 64, 128);
    conv(wf_W2, wfW2T, 128, 128, 128, 128);
    conv(wf_W3, wfW3T, 128, 1, 128, 16);
    conv(nf_W1, nfW1T, 25, 128, 32, 128);
    conv(nf_W2, nfW2T, 128, 128, 128, 128);
    conv(nf_W3, nfW3T, 128, 6, 128, 16);

    nerf_main<<<NRAYS / 2, 256, 0, stream>>>(
        ofs, results, noise, ffW1T, ffW2T, ffW3T, wfW1T, wfW2T, wfW3T,
        ff_b1, ff_b2, ff_b3, wf_b1, wf_b2, wf_b3, (float*)d_out);

    naive_kern<<<NRAYS / 128, 256, 0, stream>>>(
        x, nfW1T, nfW2T, nfW3T, nf_b1, nf_b2, nf_b3,
        nm_W1, nm_b1, nm_W2, nm_b2, nm_W3, nm_b3, (float*)d_out);
}
